// GroupMultiLabelCE_12128987644154
// MI455X (gfx1250) — compile-verified, matrix operand order checked by reading, not run on hardware
//
#include <hip/hip_runtime.h>
#include <stdint.h>

// Problem constants (fixed by the reference)
#define NUM_CLASS 21
#define NUM_SPX   2048
#define NN        8
#define HH        512
#define WW        512
#define HWPIX     (HH * WW)            // 262144 = 2^18
#define EPSV      1e-8f
#define BLK       256                  // pass-2 block
#define BLKP      1024                 // pass-1 block (32 waves on one WGP)
#define CHUNKS    16                   // pass-1 chunks per sample -> 128 blocks
#define PIX_PER_BLOCK (HWPIX / CHUNKS) // 16384
#define ITERS     (PIX_PER_BLOCK / BLKP) // 16
#define SPC       (NUM_SPX * NUM_CLASS)  // 43008 buckets per sample

typedef float    v2f  __attribute__((ext_vector_type(2)));
typedef float    v8f  __attribute__((ext_vector_type(8)));
typedef _Float16 v16h __attribute__((ext_vector_type(16)));

// ---------------------------------------------------------------------------
// CDNA5 async global->LDS path. Existence proven in round 1 (the builtin
// type-checked); signature: (as1 int* src, as3 int* dst, imm offset, imm cpol).
// ---------------------------------------------------------------------------
#if __has_builtin(__builtin_amdgcn_global_load_async_to_lds_b32) && \
    __has_builtin(__builtin_amdgcn_s_wait_asynccnt)
#define HAVE_ASYNC_LDS 1
typedef __attribute__((address_space(1))) int gint_as;   // global int
typedef __attribute__((address_space(3))) int lint_as;   // LDS int
#else
#define HAVE_ASYNC_LDS 0
#endif

// ---------------------------------------------------------------------------
// Exact wave32 sum on the WMMA pipe (v_wmma_f32_16x16x4_f32).
//   A: lane m -> {v[m],0}, lane m+16 -> {v[m+16],0};  B = ones
//   => D[m][*] = v[m] + v[m+16]; a lane's 8 accumulator VGPRs sum rows 0..7
//   (lanes 0-15) or 8..15 (lanes 16-31); one shfl_xor(16) finishes the sum.
// EXEC must be all ones at the call site.
// ---------------------------------------------------------------------------
__device__ __forceinline__ float wave_reduce_wmma(float v) {
    v8f c = {};
    v8f d;
#if __has_builtin(__builtin_amdgcn_wmma_f32_16x16x4_f32)
    v2f a; a.x = v;    a.y = 0.0f;
    v2f b; b.x = 1.0f; b.y = 1.0f;
    d = __builtin_amdgcn_wmma_f32_16x16x4_f32(false, a, false, b,
                                              (short)0, c, false, false);
#else
    v16h a = {};
    a[0] = (_Float16)v;
    v16h b;
#pragma unroll
    for (int i = 0; i < 16; ++i) b[i] = (_Float16)1.0f;
    d = __builtin_amdgcn_wmma_f32_16x16x32_f16(false, a, false, b,
                                               (short)0, c, false, false);
#endif
    float s = d[0] + d[1] + d[2] + d[3] + d[4] + d[5] + d[6] + d[7];
    s += __shfl_xor(s, 16, 32);
    return s;
}

// ---------------------------------------------------------------------------
// Zero the workspace (global max buckets + 2 accumulators) every call.
// ---------------------------------------------------------------------------
__global__ void k_zero(float* __restrict__ p, int n) {
    int i = blockIdx.x * BLK + threadIdx.x;
    if (i < n) p[i] = 0.0f;
}

// ---------------------------------------------------------------------------
// Pass 1: softmax + segment-max with LDS-privatized buckets.
//
// Atomic roofline: the naive scheme is 21 global atomics per masked pixel
// (~44M) — far above the ~8us HBM floor for the 176MB input stream. Here each
// 1024-thread block (one WGP, 32 waves) keeps a private 2048x21 f32 max-table
// in LDS (168KB) and a 21x1024 async staging tile (84KB): 252KB of the
// 320KB/WGP LDS. All per-pixel maxes become ds_max ops over 64 banks; only
// one flush of <=43008 entries per block (128 blocks => ~5.5M global atomics,
// 8x fewer) touches L2.
//
// Positive-float trick: softmax outputs are > 0, so int ordering == float
// ordering and zero-init reproduces the reference's "empty segment -> 0".
// Each thread consumes only its own LDS tile column, so s_wait_asynccnt(0)
// is the only sync needed per stage iteration.
// ---------------------------------------------------------------------------
__global__ void __launch_bounds__(BLKP)
k_softmax_spmax(const float* __restrict__ inputs,
                const int* __restrict__ superpixels,
                const unsigned char* __restrict__ spmasks,
                int* __restrict__ bucket_i) {
    __shared__ float tile[NUM_CLASS * BLKP];   // 84 KB staging
    __shared__ int   lbkt[SPC];                // 168 KB private max table

    const int tid = threadIdx.x;
    const int cx  = blockIdx.x;                // chunk within sample
    const int n   = blockIdx.y;                // sample

    // zero the private table
#pragma unroll
    for (int e = tid; e < SPC; e += BLKP) lbkt[e] = 0;
    __syncthreads();

    const float* nbase = inputs + (size_t)n * NUM_CLASS * HWPIX;

    for (int it = 0; it < ITERS; ++it) {
        const int hw = cx * PIX_PER_BLOCK + it * BLKP + tid;
        const float* gbase = nbase + hw;

#if HAVE_ASYNC_LDS
#pragma unroll
        for (int c = 0; c < NUM_CLASS; ++c) {
            __builtin_amdgcn_global_load_async_to_lds_b32(
                (gint_as*)(gbase + (size_t)c * HWPIX),
                (lint_as*)&tile[c * BLKP + tid],
                0, 0);
        }
        __builtin_amdgcn_s_wait_asynccnt(0);
#else
#pragma unroll
        for (int c = 0; c < NUM_CLASS; ++c)
            tile[c * BLKP + tid] = gbase[(size_t)c * HWPIX];
#endif

        if (spmasks[(size_t)n * HWPIX + hw]) {
            const int sp = superpixels[(size_t)n * HWPIX + hw];
            // numerically stable softmax (TEMP == 1)
            float m = -3.402823466e38f;
#pragma unroll
            for (int c = 0; c < NUM_CLASS; ++c)
                m = fmaxf(m, tile[c * BLKP + tid]);
            float ssum = 0.0f;
#pragma unroll
            for (int c = 0; c < NUM_CLASS; ++c) {
                float e = __expf(tile[c * BLKP + tid] - m);
                tile[c * BLKP + tid] = e;      // thread-private slot
                ssum += e;
            }
            const float inv = 1.0f / ssum;
            int* lb = lbkt + sp * NUM_CLASS;
#pragma unroll
            for (int c = 0; c < NUM_CLASS; ++c) {
                float p = tile[c * BLKP + tid] * inv;   // p > 0 always
                atomicMax(&lb[c], __float_as_int(p));   // ds_max, bank-parallel
            }
        }
    }

    __syncthreads();
    // flush private table to the global max table (skip empty entries)
    int* gb = bucket_i + (size_t)n * SPC;
#pragma unroll 4
    for (int e = tid; e < SPC; e += BLKP) {
        int v = lbkt[e];
        if (v != 0) atomicMax(&gb[e], v);
    }
}

// ---------------------------------------------------------------------------
// Pass 2: per-superpixel loss over the 8x2048x21 table, wave-reduced on the
// WMMA pipe, then one atomicAdd per wave into {loss_sum, valid_count}.
// Grid covers exactly N*NUM_SPX threads -> EXEC all-ones at the WMMA.
// ---------------------------------------------------------------------------
__global__ void k_sp_loss(const float* __restrict__ targets,
                          const float* __restrict__ mx,
                          float* __restrict__ acc) {
    const int idx = blockIdx.x * BLK + threadIdx.x;      // 0 .. N*NUM_SPX-1
    const float* t = targets + (size_t)idx * (NUM_CLASS + 1);
    const float* m = mx + (size_t)idx * NUM_CLASS;

    float loss = 0.0f, cnt = 0.0f;
    bool  any  = false;
    float trg[NUM_CLASS];
#pragma unroll
    for (int c = 0; c < NUM_CLASS; ++c) {
        trg[c] = t[c];
        any = any | (trg[c] != 0.0f);
    }
    if (any) {
#pragma unroll
        for (int c = 0; c < NUM_CLASS; ++c) {
            float top = m[c] * trg[c];
            if (top != 0.0f) {                 // implies empty_trg_mask too
                loss += -__logf(top + EPSV);
                cnt  += 1.0f;
            }
        }
    }
    // converged here: full-wave WMMA reductions
    float ls = wave_reduce_wmma(loss);
    float cs = wave_reduce_wmma(cnt);
    if ((threadIdx.x & 31) == 0) {
        atomicAdd(&acc[0], ls);
        atomicAdd(&acc[1], cs);
    }
}

__global__ void k_finalize(const float* __restrict__ acc,
                           float* __restrict__ out) {
    out[0] = acc[0] / (1.0f + acc[1]);         // num_valid = 1 + valid.sum()
}

// ---------------------------------------------------------------------------
extern "C" void kernel_launch(void* const* d_in, const int* in_sizes, int n_in,
                              void* d_out, int out_size, void* d_ws, size_t ws_size,
                              hipStream_t stream) {
    const float*         inputs      = (const float*)d_in[0];
    const float*         targets     = (const float*)d_in[1];
    const int*           superpixels = (const int*)d_in[2];
    const unsigned char* spmasks     = (const unsigned char*)d_in[3];

    float* ws     = (float*)d_ws;
    float* bucket = ws;                          // N*SPC floats
    float* acc    = ws + (size_t)NN * SPC;       // {loss_sum, valid_count}

    const int nz = NN * SPC + 2;
    k_zero<<<(nz + BLK - 1) / BLK, BLK, 0, stream>>>(ws, nz);

    dim3 g1(CHUNKS, NN);
    k_softmax_spmax<<<g1, BLKP, 0, stream>>>(inputs, superpixels, spmasks,
                                             (int*)bucket);

    k_sp_loss<<<(NN * NUM_SPX) / BLK, BLK, 0, stream>>>(targets, bucket, acc);

    k_finalize<<<1, 1, 0, stream>>>(acc, (float*)d_out);
}
